// DeepSeekLayer_64278480552140
// MI455X (gfx1250) — matrix operand order repaired and checked by reading, not compiled, not run on hardware
//
#include <hip/hip_runtime.h>
#include <hip/hip_bf16.h>

// ---------------------------------------------------------------------------
// DeepSeek-style layer for MI455X (gfx1250), wave32 + WMMA bf16.
// Dense math: v_wmma_f32_16x16x32_bf16, f32 accumulate. Weights pre-transposed
// to bf16 [N,K]; activations bf16; all operand fetches 16B (b128).
// GEMMs: double-buffered LDS, 128x64 block tile, 32x64 wave tile.
// Attention: flash-style, V stored transposed bf16.
// ---------------------------------------------------------------------------

typedef __attribute__((ext_vector_type(8)))  __bf16 v8bf;
typedef __attribute__((ext_vector_type(16))) __bf16 v16bf;
typedef __attribute__((ext_vector_type(8)))  float  v8f;

#define B_SZ    2
#define S_SZ    2048
#define D_SZ    1024
#define H_SZ    16
#define DH_SZ   64
#define DR_SZ   64
#define DHID_SZ 512
#define NR_SZ   8
#define T_SZ    (B_SZ * S_SZ)   // 4096 tokens
#define LDSP    40              // LDS row stride (bf16) -> 80B, 16B aligned

__device__ __forceinline__ v8f wmma_bf16(v16bf a, v16bf b, v8f c) {
  return __builtin_amdgcn_wmma_f32_16x16x32_bf16(
      false, a, false, b, (short)0, c, false, false);
}

__device__ __forceinline__ v16bf cat16(v8bf lo, v8bf hi) {
  v16bf r;
#pragma unroll
  for (int j = 0; j < 8; ++j) { r[j] = lo[j]; r[8 + j] = hi[j]; }
  return r;
}

// A fragment (16x32): lanes 0-15 K=0..15, lanes 16-31 K=16..31 (contiguous).
__device__ __forceinline__ v16bf a_frag_bf(const __bf16* __restrict__ rowp, int lane) {
  const int kb = (lane & 16) ? 16 : 0;
  return cat16(*(const v8bf*)(rowp + kb), *(const v8bf*)(rowp + kb + 8));
}
// B fragment (32x16): lanes 0-15 K=0..15, lanes 16-31 K=16..31 (contiguous).
__device__ __forceinline__ v16bf b_frag_bf(const __bf16* __restrict__ rowp, int lane) {
  const int kh = (lane & 16) ? 16 : 0;
  return cat16(*(const v8bf*)(rowp + kh), *(const v8bf*)(rowp + kh + 8));
}
// A fragment from fp32 memory (16 contiguous per half-wave).
__device__ __forceinline__ v16bf a_frag_f32(const float* __restrict__ p) {
  float4 f0 = *(const float4*)(p);
  float4 f1 = *(const float4*)(p + 4);
  float4 f2 = *(const float4*)(p + 8);
  float4 f3 = *(const float4*)(p + 12);
  v16bf r;
  r[0]=(__bf16)f0.x; r[1]=(__bf16)f0.y; r[2]=(__bf16)f0.z; r[3]=(__bf16)f0.w;
  r[4]=(__bf16)f1.x; r[5]=(__bf16)f1.y; r[6]=(__bf16)f1.z; r[7]=(__bf16)f1.w;
  r[8]=(__bf16)f2.x; r[9]=(__bf16)f2.y; r[10]=(__bf16)f2.z; r[11]=(__bf16)f2.w;
  r[12]=(__bf16)f3.x; r[13]=(__bf16)f3.y; r[14]=(__bf16)f3.z; r[15]=(__bf16)f3.w;
  return r;
}
// B fragment from 16 contiguous fp32.
__device__ __forceinline__ v16bf b_frag_f32(const float* __restrict__ p) {
  float4 f0 = *(const float4*)(p);
  float4 f1 = *(const float4*)(p + 4);
  float4 f2 = *(const float4*)(p + 8);
  float4 f3 = *(const float4*)(p + 12);
  v16bf r;
  r[0]=(__bf16)f0.x; r[1]=(__bf16)f0.y; r[2]=(__bf16)f0.z; r[3]=(__bf16)f0.w;
  r[4]=(__bf16)f1.x; r[5]=(__bf16)f1.y; r[6]=(__bf16)f1.z; r[7]=(__bf16)f1.w;
  r[8]=(__bf16)f2.x; r[9]=(__bf16)f2.y; r[10]=(__bf16)f2.z; r[11]=(__bf16)f2.w;
  r[12]=(__bf16)f3.x; r[13]=(__bf16)f3.y; r[14]=(__bf16)f3.z; r[15]=(__bf16)f3.w;
  return r;
}

// ---------------------------------------------------------------------------
// Weight transpose + fp32->bf16: in[E,K,N] -> out[E,N,K] bf16.
// ---------------------------------------------------------------------------
__global__ void transpose_to_bf16(const float* __restrict__ in,
                                  __bf16* __restrict__ outp,
                                  int E, int K, int N) {
  size_t idx = (size_t)blockIdx.x * blockDim.x + threadIdx.x;
  size_t total = (size_t)E * K * N;
  if (idx >= total) return;
  size_t e = idx / ((size_t)K * N);
  size_t r = idx - e * (size_t)K * N;
  size_t k = r / N, n = r - k * N;
  outp[e * (size_t)N * K + n * K + k] = (__bf16)in[idx];
}

// ---------------------------------------------------------------------------
// RMSNorm: fp32 in -> bf16 out.
// ---------------------------------------------------------------------------
__global__ void rmsnorm_kernel(const float* __restrict__ x,
                               const float* __restrict__ w,
                               __bf16* __restrict__ out, int D) {
  const int t = blockIdx.x;
  const float* xp = x + (size_t)t * D;
  float ss = 0.f;
  for (int d = threadIdx.x; d < D; d += blockDim.x) { float v = xp[d]; ss += v * v; }
  __shared__ float red[256];
  red[threadIdx.x] = ss;
  __syncthreads();
  for (int st = 128; st > 0; st >>= 1) {
    if ((int)threadIdx.x < st) red[threadIdx.x] += red[threadIdx.x + st];
    __syncthreads();
  }
  float scale = rsqrtf(red[0] / (float)D + 1e-6f);
  for (int d = threadIdx.x; d < D; d += blockDim.x)
    out[(size_t)t * D + d] = (__bf16)(xp[d] * scale * w[d]);
}

// ---------------------------------------------------------------------------
// Double-buffered WMMA GEMM: C = A[M,K](bf16) x Bt[N,K](bf16).
// Block 128 = 4 waves; block tile 128x64; wave tile 32x64; K-step 32.
// mode: 0 = f32 (+res), 1 = bf16, 2 = bf16 transposed [N,M],
//       3 = f32 accumulate out += scale(row) * acc (scales nullable, stride 8).
// ---------------------------------------------------------------------------
__global__ void wmma_gemm_bt(const __bf16* __restrict__ A,
                             const __bf16* __restrict__ Bt,
                             const float* __restrict__ res,
                             const float* __restrict__ scales,
                             void* __restrict__ Cout,
                             int M, int N, int K, int mode) {
  __shared__ __align__(16) __bf16 aLds[2][128][LDSP];
  __shared__ __align__(16) __bf16 bLds[2][64][LDSP];
  const int tid = threadIdx.x;
  const int lane = tid & 31;
  const int wave = tid >> 5;
  const int ln = lane & 15;
  const int hi = (lane & 16) ? 8 : 0;
  const int bM = blockIdx.y * 128;
  const int bN = blockIdx.x * 64;
  const int rowB = tid >> 1;
  const int halfB = (tid & 1) * 16;
  const __bf16* gA = A + (size_t)(bM + tid) * K;       // 1 row (32 bf16) / thread
  const __bf16* gB = Bt + (size_t)(bN + rowB) * K + halfB;

  auto stage = [&](int k0, int buf) {
    v8bf a0 = *(const v8bf*)(gA + k0);
    v8bf a1 = *(const v8bf*)(gA + k0 + 8);
    v8bf a2 = *(const v8bf*)(gA + k0 + 16);
    v8bf a3 = *(const v8bf*)(gA + k0 + 24);
    v8bf b0 = *(const v8bf*)(gB + k0);
    v8bf b1 = *(const v8bf*)(gB + k0 + 8);
    *(v8bf*)&aLds[buf][tid][0]  = a0;
    *(v8bf*)&aLds[buf][tid][8]  = a1;
    *(v8bf*)&aLds[buf][tid][16] = a2;
    *(v8bf*)&aLds[buf][tid][24] = a3;
    *(v8bf*)&bLds[buf][rowB][halfB]     = b0;
    *(v8bf*)&bLds[buf][rowB][halfB + 8] = b1;
  };

  v8f zero = {0, 0, 0, 0, 0, 0, 0, 0};
  v8f acc[2][4] = {{zero, zero, zero, zero}, {zero, zero, zero, zero}};

  stage(0, 0);
  for (int k0 = 0; k0 < K; k0 += 32) {
    const int buf = (k0 >> 5) & 1;
    __syncthreads();                       // buf staged; prev reads done
    if (k0 + 32 < K) stage(k0 + 32, buf ^ 1);
    v16bf a0 = a_frag_bf(&aLds[buf][wave * 32 + ln][0], lane);
    v16bf a1 = a_frag_bf(&aLds[buf][wave * 32 + 16 + ln][0], lane);
#pragma unroll
    for (int t = 0; t < 4; ++t) {
      v16bf b = b_frag_bf(&bLds[buf][t * 16 + ln][0], lane);
      acc[0][t] = wmma_bf16(a0, b, acc[0][t]);
      acc[1][t] = wmma_bf16(a1, b, acc[1][t]);
    }
  }

#pragma unroll
  for (int sub = 0; sub < 2; ++sub) {
#pragma unroll
    for (int i = 0; i < 8; ++i) {
      const int r = bM + wave * 32 + sub * 16 + i + hi;
      if (mode == 0) {
        float* C = (float*)Cout;
#pragma unroll
        for (int t = 0; t < 4; ++t) {
          size_t idx = (size_t)r * N + (bN + t * 16 + ln);
          float v = acc[sub][t][i];
          if (res) v += res[idx];
          C[idx] = v;
        }
      } else if (mode == 1) {
        __bf16* C = (__bf16*)Cout;
#pragma unroll
        for (int t = 0; t < 4; ++t)
          C[(size_t)r * N + (bN + t * 16 + ln)] = (__bf16)acc[sub][t][i];
      } else if (mode == 2) {
        __bf16* C = (__bf16*)Cout;  // [N, M]
#pragma unroll
        for (int t = 0; t < 4; ++t)
          C[(size_t)(bN + t * 16 + ln) * M + r] = (__bf16)acc[sub][t][i];
      } else {  // mode 3: scaled accumulate
        float* C = (float*)Cout;
        float sc = scales ? scales[(size_t)r * NR_SZ] : 1.0f;
#pragma unroll
        for (int t = 0; t < 4; ++t) {
          size_t idx = (size_t)r * N + (bN + t * 16 + ln);
          C[idx] += sc * acc[sub][t][i];
        }
      }
    }
  }
}

// ---------------------------------------------------------------------------
// RoPE in place (fp32).
// ---------------------------------------------------------------------------
__global__ void rope_kernel(float* __restrict__ buf, int T, int G,
                            int rowStride, int groupStride, int offset,
                            int Slen) {
  int idx = blockIdx.x * blockDim.x + threadIdx.x;
  if (idx >= T * G * 32) return;
  int i = idx & 31;
  int g = (idx >> 5) % G;
  int t = idx / (32 * G);
  int s = t % Slen;
  float inv = __expf(-(float)i * (9.210340371976184f / 32.f));  // 10000^(-i/32)
  float ang = (float)s * inv;
  float cs = cosf(ang), sn = sinf(ang);
  float* p = buf + (size_t)t * rowStride + (size_t)g * groupStride + offset;
  float t1 = p[i], t2 = p[i + 32];
  p[i] = t1 * cs - t2 * sn;
  p[i + 32] = t2 * cs + t1 * sn;
}

// ---------------------------------------------------------------------------
// Flash attention. grid (S/16, H, B), 1 wave/block, 16 queries/wave.
// q fp32 [T,H*128]; kc bf16 [T,H*64]; kr fp32 [T,64]; vt bf16 [H*64,T].
// ---------------------------------------------------------------------------
__global__ void flash_attn_kernel(const float* __restrict__ qbuf,
                                  const __bf16* __restrict__ kc,
                                  const float* __restrict__ kr,
                                  const __bf16* __restrict__ vt,
                                  __bf16* __restrict__ ctx) {
  const int lane = threadIdx.x;
  const int ln = lane & 15;
  const int hi = (lane & 16) ? 8 : 0;
  const int qt = blockIdx.x, h = blockIdx.y, b = blockIdx.z;
  const int bS = b * S_SZ;
  const int row0 = bS + qt * 16;

  __shared__ __align__(16) __bf16 p_lds[16][LDSP];

  v16bf qf[4];
  {
    const float* qrow = qbuf + (size_t)(row0 + ln) * (H_SZ * 128) + h * 128;
#pragma unroll
    for (int c = 0; c < 4; ++c)
      qf[c] = a_frag_f32(qrow + c * 32 + ((lane & 16) ? 16 : 0));
  }

  v8f zero = {0, 0, 0, 0, 0, 0, 0, 0};
  v8f o[4] = {zero, zero, zero, zero};
  float mrow[8], lrow[8], corr[8];
#pragma unroll
  for (int i = 0; i < 8; ++i) { mrow[i] = -1e30f; lrow[i] = 0.f; }

  const float scale = 0.08838834764831845f;  // 1/sqrt(128)
  const int qmax = qt * 16 + 15;

  for (int kb = 0; kb <= qmax; kb += 32) {
    v8f s0 = zero, s1 = zero;
#pragma unroll
    for (int c = 0; c < 4; ++c) {
      const int dbase = c * 32 + ((lane & 16) ? 16 : 0);
#pragma unroll
      for (int half = 0; half < 2; ++half) {
        const int key = bS + kb + half * 16 + ln;
        v16bf f;
        if (c < 2) {
          const __bf16* p = kc + (size_t)key * (H_SZ * DH_SZ) + h * DH_SZ + dbase;
          f = cat16(*(const v8bf*)p, *(const v8bf*)(p + 8));
        } else {
          f = b_frag_f32(kr + (size_t)key * DR_SZ + (dbase - 64));
        }
        if (half == 0) s0 = wmma_bf16(qf[c], f, s0);
        else           s1 = wmma_bf16(qf[c], f, s1);
      }
    }
#pragma unroll
    for (int i = 0; i < 8; ++i) {
      int qrow = qt * 16 + i + hi;
      float v0 = s0[i] * scale;
      float v1 = s1[i] * scale;
      if (kb + ln > qrow) v0 = -1e9f;
      if (kb + 16 + ln > qrow) v1 = -1e9f;
      float mx = fmaxf(v0, v1);
      mx = fmaxf(mx, __shfl_xor(mx, 1, 32));
      mx = fmaxf(mx, __shfl_xor(mx, 2, 32));
      mx = fmaxf(mx, __shfl_xor(mx, 4, 32));
      mx = fmaxf(mx, __shfl_xor(mx, 8, 32));
      float mnew = fmaxf(mrow[i], mx);
      float p0 = __expf(v0 - mnew);
      float p1 = __expf(v1 - mnew);
      float rs = p0 + p1;
      rs += __shfl_xor(rs, 1, 32);
      rs += __shfl_xor(rs, 2, 32);
      rs += __shfl_xor(rs, 4, 32);
      rs += __shfl_xor(rs, 8, 32);
      corr[i] = __expf(mrow[i] - mnew);
      lrow[i] = lrow[i] * corr[i] + rs;
      mrow[i] = mnew;
      p_lds[i + hi][ln] = (__bf16)p0;
      p_lds[i + hi][16 + ln] = (__bf16)p1;
    }
#pragma unroll
    for (int c2 = 0; c2 < 4; ++c2)
#pragma unroll
      for (int i = 0; i < 8; ++i) o[c2][i] *= corr[i];
    __syncthreads();

    v16bf pf = a_frag_bf(&p_lds[ln][0], lane);
    const int khi = (lane & 16) ? 16 : 0;
#pragma unroll
    for (int c2 = 0; c2 < 4; ++c2) {
      const __bf16* vp = vt + (size_t)(h * DH_SZ + c2 * 16 + ln) * T_SZ +
                         (bS + kb + khi);
      v16bf vf = cat16(*(const v8bf*)vp, *(const v8bf*)(vp + 8));
      o[c2] = wmma_bf16(pf, vf, o[c2]);
    }
    __syncthreads();
  }

#pragma unroll
  for (int i = 0; i < 8; ++i) {
    float inv = 1.0f / lrow[i];
    size_t base = (size_t)(row0 + i + hi) * (H_SZ * DH_SZ) + h * DH_SZ;
#pragma unroll
    for (int c2 = 0; c2 < 4; ++c2)
      ctx[base + c2 * 16 + ln] = (__bf16)(o[c2][i] * inv);
  }
}

// ---------------------------------------------------------------------------
// Gating: one wave/token; softmax + top-2 -> dw[T,8].
// ---------------------------------------------------------------------------
__global__ void gate_topk_kernel(const __bf16* __restrict__ h,
                                 const float* __restrict__ gw,
                                 float* __restrict__ dw, int T) {
  const int wave = threadIdx.x >> 5;
  const int lane = threadIdx.x & 31;
  const int t = blockIdx.x * (blockDim.x >> 5) + wave;
  if (t >= T) return;
  float acc[NR_SZ];
#pragma unroll
  for (int e = 0; e < NR_SZ; ++e) acc[e] = 0.f;
  const __bf16* hp = h + (size_t)t * D_SZ;
  for (int d = lane; d < D_SZ; d += 32) {
    float hv = (float)hp[d];
#pragma unroll
    for (int e = 0; e < NR_SZ; ++e) acc[e] += hv * gw[d * NR_SZ + e];
  }
#pragma unroll
  for (int e = 0; e < NR_SZ; ++e)
    for (int m = 16; m >= 1; m >>= 1) acc[e] += __shfl_xor(acc[e], m, 32);
  if (lane == 0) {
    float mx = acc[0];
#pragma unroll
    for (int e = 1; e < NR_SZ; ++e) mx = fmaxf(mx, acc[e]);
    float ex[NR_SZ], s = 0.f;
#pragma unroll
    for (int e = 0; e < NR_SZ; ++e) { ex[e] = __expf(acc[e] - mx); s += ex[e]; }
#pragma unroll
    for (int e = 0; e < NR_SZ; ++e) ex[e] /= s;
    int i0 = 0;
#pragma unroll
    for (int e = 1; e < NR_SZ; ++e) if (ex[e] > ex[i0]) i0 = e;
    int i1 = (i0 == 0) ? 1 : 0;
#pragma unroll
    for (int e = 0; e < NR_SZ; ++e) if (e != i0 && ex[e] > ex[i1]) i1 = e;
    float gs = ex[i0] + ex[i1];
#pragma unroll
    for (int e = 0; e < NR_SZ; ++e)
      dw[(size_t)t * NR_SZ + e] =
          (e == i0) ? ex[i0] / gs : ((e == i1) ? ex[i1] / gs : 0.f);
  }
}

// ---------------------------------------------------------------------------
// Fused expert gate/up: GU(bf16) = silu(A@Wg) * (A@Wu). Double-buffered LDS,
// block tile 64x64, wave tile 16x64, shared A fragment across both products.
// ---------------------------------------------------------------------------
__global__ void moe_gateup_kernel(const __bf16* __restrict__ A,
                                  const __bf16* __restrict__ Wg,  // [N,K]
                                  const __bf16* __restrict__ Wu,  // [N,K]
                                  __bf16* __restrict__ GU,
                                  int M, int N, int K) {
  __shared__ __align__(16) __bf16 aLds[2][64][LDSP];
  __shared__ __align__(16) __bf16 gLds[2][64][LDSP];
  __shared__ __align__(16) __bf16 uLds[2][64][LDSP];
  const int tid = threadIdx.x;
  const int lane = tid & 31;
  const int wave = tid >> 5;
  const int ln = lane & 15;
  const int hi = (lane & 16) ? 8 : 0;
  const int bM = blockIdx.y * 64;
  const int bN = blockIdx.x * 64;
  const int row = tid >> 1;
  const int half = (tid & 1) * 16;
  const __bf16* gA = A + (size_t)(bM + row) * K + half;
  const __bf16* gG = Wg + (size_t)(bN + row) * K + half;
  const __bf16* gU = Wu + (size_t)(bN + row) * K + half;

  auto stage = [&](int k0, int buf) {
    v8bf a0 = *(const v8bf*)(gA + k0);
    v8bf a1 = *(const v8bf*)(gA + k0 + 8);
    v8bf g0 = *(const v8bf*)(gG + k0);
    v8bf g1 = *(const v8bf*)(gG + k0 + 8);
    v8bf u0 = *(const v8bf*)(gU + k0);
    v8bf u1 = *(const v8bf*)(gU + k0 + 8);
    *(v8bf*)&aLds[buf][row][half]     = a0;
    *(v8bf*)&aLds[buf][row][half + 8] = a1;
    *(v8bf*)&gLds[buf][row][half]     = g0;
    *(v8bf*)&gLds[buf][row][half + 8] = g1;
    *(v8bf*)&uLds[buf][row][half]     = u0;
    *(v8bf*)&uLds[buf][row][half + 8] = u1;
  };

  v8f zero = {0, 0, 0, 0, 0, 0, 0, 0};
  v8f cg[4] = {zero, zero, zero, zero};
  v8f cu[4] = {zero, zero, zero, zero};

  stage(0, 0);
  for (int k0 = 0; k0 < K; k0 += 32) {
    const int buf = (k0 >> 5) & 1;
    __syncthreads();
    if (k0 + 32 < K) stage(k0 + 32, buf ^ 1);
    v16bf a = a_frag_bf(&aLds[buf][wave * 16 + ln][0], lane);
#pragma unroll
    for (int t = 0; t < 4; ++t) {
      cg[t] = wmma_bf16(a, b_frag_bf(&gLds[buf][t * 16 + ln][0], lane), cg[t]);
      cu[t] = wmma_bf16(a, b_frag_bf(&uLds[buf][t * 16 + ln][0], lane), cu[t]);
    }
  }
#pragma unroll
  for (int t = 0; t < 4; ++t)
#pragma unroll
    for (int i = 0; i < 8; ++i) {
      float g = cg[t][i], u = cu[t][i];
      float sg = g / (1.0f + __expf(-g));  // silu
      GU[(size_t)(bM + wave * 16 + i + hi) * N + (bN + t * 16 + ln)] =
          (__bf16)(sg * u);
    }
}

// ---------------------------------------------------------------------------
// Launch sequence
// ---------------------------------------------------------------------------
extern "C" void kernel_launch(void* const* d_in, const int* in_sizes, int n_in,
                              void* d_out, int out_size, void* d_ws, size_t ws_size,
                              hipStream_t stream) {
  (void)in_sizes; (void)n_in; (void)out_size; (void)ws_size;
  const float* x       = (const float*)d_in[0];
  const float* w_q     = (const float*)d_in[2];   // [1024, 2048]
  const float* w_dkv   = (const float*)d_in[3];   // [1024, 512]
  const float* w_uk    = (const float*)d_in[4];   // [512, 1024]
  const float* w_uv    = (const float*)d_in[5];   // [512, 1024]
  const float* w_kr    = (const float*)d_in[6];   // [1024, 64]
  const float* w_o     = (const float*)d_in[7];   // [1024, 1024]
  const float* attn_nw = (const float*)d_in[8];
  const float* mlp_nw  = (const float*)d_in[9];
  const float* gate_w  = (const float*)d_in[10];  // [1024, 8]
  const float* wr_gate = (const float*)d_in[11];  // [8, 1024, 512]
  const float* wr_up   = (const float*)d_in[12];
  const float* wr_down = (const float*)d_in[13];  // [8, 512, 1024]
  const float* ws_gate = (const float*)d_in[14];  // [2, 1024, 512]
  const float* ws_up   = (const float*)d_in[15];
  const float* ws_down = (const float*)d_in[16];  // [2, 512, 1024]
  float* out = (float*)d_out;

  const int T = T_SZ;
  char* p = (char*)d_ws;
  auto alloc = [&](size_t bytes) {
    char* r = p;
    p += (bytes + 255) & ~(size_t)255;
    return r;
  };
  __bf16* hbuf   = (__bf16*)alloc((size_t)T * 1024 * 2);
  float*  qbuf   = (float*) alloc((size_t)T * 2048 * 4);
  __bf16* latent = (__bf16*)alloc((size_t)T * 512 * 2);
  __bf16* kcbuf  = (__bf16*)alloc((size_t)T * 1024 * 2);
  __bf16* vtbuf  = (__bf16*)alloc((size_t)1024 * T * 2);
  float*  krbuf  = (float*) alloc((size_t)T * 64 * 4);
  __bf16* ctxbuf = (__bf16*)alloc((size_t)T * 1024 * 2);
  float*  dw     = (float*) alloc((size_t)T * NR_SZ * 4);
  __bf16* gu     = (__bf16*)alloc((size_t)T * 512 * 2);
  __bf16* wqT    = (__bf16*)alloc((size_t)2048 * 1024 * 2);
  __bf16* wdkvT  = (__bf16*)alloc((size_t)512 * 1024 * 2);
  __bf16* wukT   = (__bf16*)alloc((size_t)1024 * 512 * 2);
  __bf16* wuvT   = (__bf16*)alloc((size_t)1024 * 512 * 2);
  __bf16* wkrT   = (__bf16*)alloc((size_t)64 * 1024 * 2);
  __bf16* woT    = (__bf16*)alloc((size_t)1024 * 1024 * 2);
  __bf16* wrgT   = (__bf16*)alloc((size_t)8 * 512 * 1024 * 2);
  __bf16* wruT   = (__bf16*)alloc((size_t)8 * 512 * 1024 * 2);
  __bf16* wrdT   = (__bf16*)alloc((size_t)8 * 1024 * 512 * 2);
  __bf16* wsgT   = (__bf16*)alloc((size_t)2 * 512 * 1024 * 2);
  __bf16* wsuT   = (__bf16*)alloc((size_t)2 * 512 * 1024 * 2);
  __bf16* wsdT   = (__bf16*)alloc((size_t)2 * 1024 * 512 * 2);

  auto tl = [&](const float* in, __bf16* o, int E, int K, int N) {
    size_t total = (size_t)E * K * N;
    transpose_to_bf16<<<(unsigned)((total + 255) / 256), 256, 0, stream>>>(in, o, E, K, N);
  };
  tl(w_q, wqT, 1, 1024, 2048);
  tl(w_dkv, wdkvT, 1, 1024, 512);
  tl(w_uk, wukT, 1, 512, 1024);
  tl(w_uv, wuvT, 1, 512, 1024);
  tl(w_kr, wkrT, 1, 1024, 64);
  tl(w_o, woT, 1, 1024, 1024);
  tl(wr_gate, wrgT, 8, 1024, 512);
  tl(wr_up, wruT, 8, 1024, 512);
  tl(wr_down, wrdT, 8, 512, 1024);
  tl(ws_gate, wsgT, 2, 1024, 512);
  tl(ws_up, wsuT, 2, 1024, 512);
  tl(ws_down, wsdT, 2, 512, 1024);

  dim3 blk(128);

  // ---- attention branch ----
  rmsnorm_kernel<<<T, 256, 0, stream>>>(x, attn_nw, hbuf, D_SZ);
  wmma_gemm_bt<<<dim3(2048 / 64, T / 128), blk, 0, stream>>>(hbuf, wqT, nullptr, nullptr, qbuf, T, 2048, 1024, 0);
  wmma_gemm_bt<<<dim3(512 / 64, T / 128), blk, 0, stream>>>(hbuf, wdkvT, nullptr, nullptr, latent, T, 512, 1024, 1);
  wmma_gemm_bt<<<dim3(1024 / 64, T / 128), blk, 0, stream>>>(latent, wukT, nullptr, nullptr, kcbuf, T, 1024, 512, 1);
  wmma_gemm_bt<<<dim3(1024 / 64, T / 128), blk, 0, stream>>>(latent, wuvT, nullptr, nullptr, vtbuf, T, 1024, 512, 2);
  wmma_gemm_bt<<<dim3(64 / 64, T / 128), blk, 0, stream>>>(hbuf, wkrT, nullptr, nullptr, krbuf, T, 64, 1024, 0);
  rope_kernel<<<(T * H_SZ * 32) / 256, 256, 0, stream>>>(qbuf, T, H_SZ, 2048, 128, DH_SZ, S_SZ);
  rope_kernel<<<(T * 32) / 256, 256, 0, stream>>>(krbuf, T, 1, DR_SZ, DR_SZ, 0, S_SZ);
  flash_attn_kernel<<<dim3(S_SZ / 16, H_SZ, B_SZ), 32, 0, stream>>>(qbuf, kcbuf, krbuf, vtbuf, ctxbuf);
  wmma_gemm_bt<<<dim3(1024 / 64, T / 128), blk, 0, stream>>>(ctxbuf, woT, x, nullptr, out, T, 1024, 1024, 0);

  // ---- MoE branch ----
  rmsnorm_kernel<<<T, 256, 0, stream>>>(out, mlp_nw, hbuf, D_SZ);
  gate_topk_kernel<<<T / 8, 256, 0, stream>>>(hbuf, gate_w, dw, T);
  for (int e = 0; e < NR_SZ; ++e) {
    moe_gateup_kernel<<<dim3(512 / 64, T / 64), blk, 0, stream>>>(
        hbuf, wrgT + (size_t)e * 512 * 1024, wruT + (size_t)e * 512 * 1024,
        gu, T, DHID_SZ, D_SZ);
    wmma_gemm_bt<<<dim3(1024 / 64, T / 128), blk, 0, stream>>>(
        gu, wrdT + (size_t)e * 1024 * 512, nullptr, dw + e, out, T, D_SZ, DHID_SZ, 3);
  }
  for (int s2 = 0; s2 < 2; ++s2) {
    moe_gateup_kernel<<<dim3(512 / 64, T / 64), blk, 0, stream>>>(
        hbuf, wsgT + (size_t)s2 * 512 * 1024, wsuT + (size_t)s2 * 512 * 1024,
        gu, T, DHID_SZ, D_SZ);
    wmma_gemm_bt<<<dim3(1024 / 64, T / 128), blk, 0, stream>>>(
        gu, wsdT + (size_t)s2 * 1024 * 512, nullptr, nullptr, out, T, D_SZ, DHID_SZ, 3);
  }
}
